// VQ_VAE1_38500086841597
// MI455X (gfx1250) — matrix operand order repaired and checked by reading, not compiled, not run on hardware
//
#include <hip/hip_runtime.h>
#include <hip/hip_bf16.h>
#include <math.h>

typedef __attribute__((ext_vector_type(16))) __bf16 v16bf;
typedef __attribute__((ext_vector_type(8)))  float  v8f;

#define EPS 1e-3f

// ---------------------------------------------------------------------------
// Encoder conv1: 4x4 s2, CI=1 -> CO=32, relu then BN.  Direct VALU kernel.
// ---------------------------------------------------------------------------
__global__ void e1_kernel(const float* __restrict__ x, const float* __restrict__ w,
                          const float* __restrict__ bias,
                          const float* __restrict__ gamma, const float* __restrict__ beta,
                          const float* __restrict__ mean,  const float* __restrict__ var,
                          float* __restrict__ out)
{
    // out: [64,64,64,32], in: [64,128,128,1], pad=1
    int idx = blockIdx.x * blockDim.x + threadIdx.x;    // 8388608 threads
    int c  =  idx        & 31;
    int ox = (idx >> 5)  & 63;
    int oy = (idx >> 11) & 63;
    int b  =  idx >> 17;
    float sum = 0.f;
    #pragma unroll
    for (int ky = 0; ky < 4; ++ky) {
        int iy = oy * 2 - 1 + ky;
        if (iy < 0 || iy >= 128) continue;
        #pragma unroll
        for (int kx = 0; kx < 4; ++kx) {
            int ix = ox * 2 - 1 + kx;
            if (ix < 0 || ix >= 128) continue;
            sum += x[(b * 128 + iy) * 128 + ix] * w[(ky * 4 + kx) * 32 + c];
        }
    }
    float v = fmaxf(sum + bias[c], 0.f);
    v = (v - mean[c]) * __frsqrt_rn(var[c] + EPS) * gamma[c] + beta[c];
    out[idx] = v;
}

// ---------------------------------------------------------------------------
// Templated WMMA conv / conv-transpose.
//   Block = 8 waves; all waves share one 32-wide output-channel PAIR of
//   16-col tiles (tnPair).  The block stages both weight slices into LDS as
//   bf16, PRE-SWIZZLED into the ISA 7.12.2 B-matrix per-lane layout, so each
//   inner-loop B operand is one contiguous 32B LDS read.  Each wave computes
//   a 16x32 f32 tile: one A-chunk (global b128 loads + packed bf16 cvt) feeds
//   TWO v_wmma_f32_16x16x32_bf16, fully unrolled over taps x K-chunks.
// ACT: 0 = relu, 1 = leaky-relu(0.3).  HAS_BN: inference batchnorm fused.
// ---------------------------------------------------------------------------
template<int IH, int IW, int CI, int OH, int OW, int CO,
         int KH, int KW, int S, int PAD, int DECONV, int ACT, int HAS_BN>
__global__ void conv_wmma_kernel(const float* __restrict__ in, const float* __restrict__ w,
                                 const float* __restrict__ bias,
                                 const float* __restrict__ gamma, const float* __restrict__ beta,
                                 const float* __restrict__ mean,  const float* __restrict__ var,
                                 float* __restrict__ out)
{
    constexpr int NCH     = CI / 32;            // K-chunks per tap
    constexpr int CHUNKS  = KH * KW * NCH;      // total 32-K chunks
    constexpr int TILES_P = CO / 32;            // tn-pair tiles

    extern __shared__ __bf16 wlds[];            // CHUNKS * 1024 bf16 (2KB/chunk)

    const int tnPair = blockIdx.x % TILES_P;
    const int tmBase = (blockIdx.x / TILES_P) * 8;
    const int wave   = threadIdx.x >> 5;
    const int lane   = threadIdx.x & 31;

    // ---- stage both 16-col weight slices into LDS, B-lane layout -----------
    // i = chunk*1024 + slice*512 + lane_s*16 + e  holds
    //     w[tap, cb*32 + (lane_s>>4)*8 + (e&7) + (e>>3)*16,  tnPair*32 + slice*16 + (lane_s&15)]
    for (int i = threadIdx.x; i < CHUNKS * 1024; i += 256) {
        const int e      = i & 15;
        const int lane_s = (i >> 4) & 31;
        const int slice  = (i >> 9) & 1;
        const int chunk  = i >> 10;
        const int tap    = chunk / NCH;
        const int cb     = chunk % NCH;
        const int c      = cb * 32 + ((lane_s >> 4) * 8) + (e & 7) + ((e >> 3) * 16);
        const int n      = tnPair * 32 + slice * 16 + (lane_s & 15);
        wlds[i] = (__bf16)w[(size_t)(tap * CI + c) * CO + n];
    }
    __syncthreads();

    // ---- per-wave tile -----------------------------------------------------
    const int tm   = tmBase + wave;
    const int mrow = tm * 16 + (lane & 15);        // A lane -> pixel (M)
    const int ox   = mrow % OW;
    const int t1   = mrow / OW;
    const int oy   = t1 % OH;
    const int b    = t1 / OH;
    const int khalfA = (lane >> 4) * 8;            // A K-split per ISA layout
    const int nn0  = tnPair * 32 + (lane & 15);    // C/D lane -> N column (tile 0)
    const int nn1  = nn0 + 16;                     // tile 1

    v8f acc0 = {};
    v8f acc1 = {};

    #pragma unroll
    for (int ky = 0; ky < KH; ++ky) {
        #pragma unroll
        for (int kx = 0; kx < KW; ++kx) {
            int iy, ix;
            bool valid;
            if (!DECONV) {
                iy = oy * S - PAD + ky;
                ix = ox * S - PAD + kx;
                valid = (iy >= 0) & (iy < IH) & (ix >= 0) & (ix < IW);
            } else {
                const int ty = oy + PAD - ky;
                const int tx = ox + PAD - kx;
                valid = (ty >= 0) & (tx >= 0) & ((ty % S) == 0) & ((tx % S) == 0);
                iy = ty / S;
                ix = tx / S;
                valid = valid & (iy < IH) & (ix < IW);
            }
            const float* ibase = in + ((size_t)(b * IH + iy) * IW + ix) * CI;

            #pragma unroll
            for (int cb = 0; cb < NCH; ++cb) {
                // A: 16 pixels x 32 channels (this lane: 1 pixel, 16 chans)
                v16bf a;
                if (valid) {
                    const float* pa = ibase + cb * 32 + khalfA;
                    __builtin_prefetch(pa + CI, 0, 1);          // global_prefetch_b8
                    float4 f0 = *(const float4*)(pa);
                    float4 f1 = *(const float4*)(pa + 4);
                    float4 f2 = *(const float4*)(pa + 16);
                    float4 f3 = *(const float4*)(pa + 20);
                    a[0]=(__bf16)f0.x; a[1]=(__bf16)f0.y; a[2]=(__bf16)f0.z; a[3]=(__bf16)f0.w;
                    a[4]=(__bf16)f1.x; a[5]=(__bf16)f1.y; a[6]=(__bf16)f1.z; a[7]=(__bf16)f1.w;
                    a[8]=(__bf16)f2.x; a[9]=(__bf16)f2.y; a[10]=(__bf16)f2.z; a[11]=(__bf16)f2.w;
                    a[12]=(__bf16)f3.x; a[13]=(__bf16)f3.y; a[14]=(__bf16)f3.z; a[15]=(__bf16)f3.w;
                } else {
                    #pragma unroll
                    for (int e = 0; e < 16; ++e) a[e] = (__bf16)0.f;
                }

                // B: two contiguous 32B LDS reads in the exact lane layout
                const int chunk = (ky * KW + kx) * NCH + cb;
                const v16bf bb0 = *(const v16bf*)&wlds[chunk * 1024 + lane * 16];
                const v16bf bb1 = *(const v16bf*)&wlds[chunk * 1024 + 512 + lane * 16];

                acc0 = __builtin_amdgcn_wmma_f32_16x16x32_bf16(
                        false, a, false, bb0, (short)0, acc0, false, false);
                acc1 = __builtin_amdgcn_wmma_f32_16x16x32_bf16(
                        false, a, false, bb1, (short)0, acc1, false, false);
            }
        }
    }

    // ---- epilogue: bias + activation + (optional) inference BN -------------
    const float bi0 = bias[nn0];
    const float bi1 = bias[nn1];
    float scale0 = 1.f, mu0 = 0.f, bet0 = 0.f;
    float scale1 = 1.f, mu1 = 0.f, bet1 = 0.f;
    if (HAS_BN) {
        scale0 = gamma[nn0] * __frsqrt_rn(var[nn0] + EPS);
        mu0 = mean[nn0];  bet0 = beta[nn0];
        scale1 = gamma[nn1] * __frsqrt_rn(var[nn1] + EPS);
        mu1 = mean[nn1];  bet1 = beta[nn1];
    }
    const int mbase = tm * 16 + (lane >> 4) * 8;   // C/D: VGPR r -> M = r (+8 hi lanes)
    #pragma unroll
    for (int r = 0; r < 8; ++r) {
        float v0 = acc0[r] + bi0;
        v0 = (ACT == 0) ? fmaxf(v0, 0.f) : (v0 >= 0.f ? v0 : 0.3f * v0);
        if (HAS_BN) v0 = (v0 - mu0) * scale0 + bet0;
        out[(size_t)(mbase + r) * CO + nn0] = v0;

        float v1 = acc1[r] + bi1;
        v1 = (ACT == 0) ? fmaxf(v1, 0.f) : (v1 >= 0.f ? v1 : 0.3f * v1);
        if (HAS_BN) v1 = (v1 - mu1) * scale1 + bet1;
        out[(size_t)(mbase + r) * CO + nn1] = v1;
    }
}

// ---------------------------------------------------------------------------
// Vector quantizer: per row (16384 rows, D=64, K=8 codes).
// ---------------------------------------------------------------------------
__global__ void vq_kernel(const float* __restrict__ z_e, const float* __restrict__ emb,
                          float* __restrict__ z_q, int* __restrict__ idx_out,
                          float* __restrict__ accum)
{
    int n = blockIdx.x * blockDim.x + threadIdx.x;   // 16384 threads
    float z[64];
    #pragma unroll
    for (int d = 0; d < 64; ++d) z[d] = z_e[(size_t)n * 64 + d];

    int best = 0;
    float bestd = 3.4e38f;
    for (int k = 0; k < 8; ++k) {
        float s = 0.f;
        #pragma unroll
        for (int d = 0; d < 64; ++d) {
            float df = z[d] - emb[k * 64 + d];
            s += df * df;
        }
        if (s < bestd) { bestd = s; best = k; }
    }
    idx_out[n] = best;

    float s = 0.f;
    #pragma unroll
    for (int d = 0; d < 64; ++d) {
        float q = emb[best * 64 + d];
        z_q[(size_t)n * 64 + d] = q;
        float df = q - z[d];
        s += df * df;
    }

    __shared__ float red[256];
    red[threadIdx.x] = s;
    __syncthreads();
    for (int off = 128; off > 0; off >>= 1) {
        if (threadIdx.x < off) red[threadIdx.x] += red[threadIdx.x + off];
        __syncthreads();
    }
    if (threadIdx.x == 0) atomicAdd(&accum[0], red[0]);
}

// ---------------------------------------------------------------------------
// Decoder final 1x1 conv 64->1 + bias + sigmoid(sigmoid(.)) + SSE vs input.
// ---------------------------------------------------------------------------
__global__ void d4_kernel(const float* __restrict__ g3, const float* __restrict__ w4,
                          const float* __restrict__ b4, const float* __restrict__ x,
                          float* __restrict__ recon, float* __restrict__ accum)
{
    int p = blockIdx.x * blockDim.x + threadIdx.x;   // 1048576 threads
    const float* row = g3 + (size_t)p * 64;
    float sum = 0.f;
    #pragma unroll
    for (int c = 0; c < 64; ++c) sum += row[c] * w4[c];
    sum += b4[0];
    float s = 1.f / (1.f + __expf(-sum));   // sigmoid applied twice (faithful to source)
    float r = 1.f / (1.f + __expf(-s));
    recon[p] = r;
    float df = x[p] - r;

    __shared__ float red[256];
    red[threadIdx.x] = df * df;
    __syncthreads();
    for (int off = 128; off > 0; off >>= 1) {
        if (threadIdx.x < off) red[threadIdx.x] += red[threadIdx.x + off];
        __syncthreads();
    }
    if (threadIdx.x == 0) atomicAdd(&accum[1], red[0]);
}

__global__ void zero_accum_kernel(float* accum)
{
    accum[0] = 0.f;
    accum[1] = 0.f;
}

__global__ void finalize_kernel(const float* __restrict__ accum, float* __restrict__ loss_out)
{
    // vq_loss = q_latent + 0.25*e_latent ; both equal mean((q-z)^2) numerically.
    float vq = 1.25f * accum[0] / (16384.f * 64.f);
    loss_out[0] = accum[1] + vq;
}

// ---------------------------------------------------------------------------
extern "C" void kernel_launch(void* const* d_in, const int* in_sizes, int n_in,
                              void* d_out, int out_size, void* d_ws, size_t ws_size,
                              hipStream_t stream)
{
    (void)in_sizes; (void)n_in; (void)out_size; (void)ws_size;

    const float* x   = (const float*)d_in[0];
    const float* e1w = (const float*)d_in[1];  const float* e1b = (const float*)d_in[2];
    const float* g1p = (const float*)d_in[3];  const float* b1p = (const float*)d_in[4];
    const float* m1p = (const float*)d_in[5];  const float* v1p = (const float*)d_in[6];
    const float* e2w = (const float*)d_in[7];  const float* e2b = (const float*)d_in[8];
    const float* g2p = (const float*)d_in[9];  const float* b2p = (const float*)d_in[10];
    const float* m2p = (const float*)d_in[11]; const float* v2p = (const float*)d_in[12];
    const float* e3w = (const float*)d_in[13]; const float* e3b = (const float*)d_in[14];
    const float* g3p = (const float*)d_in[15]; const float* b3p = (const float*)d_in[16];
    const float* m3p = (const float*)d_in[17]; const float* v3p = (const float*)d_in[18];
    const float* e4w = (const float*)d_in[19]; const float* e4b = (const float*)d_in[20];
    const float* emb = (const float*)d_in[21];
    const float* d1w = (const float*)d_in[22]; const float* d1b = (const float*)d_in[23];
    const float* dg1 = (const float*)d_in[24]; const float* db1 = (const float*)d_in[25];
    const float* dm1 = (const float*)d_in[26]; const float* dv1 = (const float*)d_in[27];
    const float* d2w = (const float*)d_in[28]; const float* d2b = (const float*)d_in[29];
    const float* dg2 = (const float*)d_in[30]; const float* db2 = (const float*)d_in[31];
    const float* dm2 = (const float*)d_in[32]; const float* dv2 = (const float*)d_in[33];
    const float* d3w = (const float*)d_in[34]; const float* d3b = (const float*)d_in[35];
    const float* dg3 = (const float*)d_in[36]; const float* db3 = (const float*)d_in[37];
    const float* dm3 = (const float*)d_in[38]; const float* dv3 = (const float*)d_in[39];
    const float* d4w = (const float*)d_in[40]; const float* d4b = (const float*)d_in[41];

    // ---- workspace layout (lifetime-based reuse) ----
    char* ws = (char*)d_ws;
    float* accum = (float*)ws;                                   // 2 floats
    char*  p     = ws + 256;
    float* Rbig  = (float*)p;  p += (size_t)268435456;           // g3  [64,128,128,64]
    float* Rmid  = (float*)p;  p += (size_t)134217728;           // h1 then g2
    float* Rg1   = (float*)p;  p += (size_t)67108864;            // h2 then g1
    float* h3    = (float*)p;  p += (size_t)2097152;             // [64,16,16,32]
    float* z_e   = (float*)p;  p += (size_t)4194304;             // [64,16,16,64]
    float* z_q   = (float*)p;  p += (size_t)4194304;             // [64,16,16,64]

    float* h1 = Rmid;   // [64,64,64,32]   33.5 MB
    float* h2 = Rg1;    // [64,32,32,32]    8.4 MB
    float* g1 = Rg1;    // [64,32,32,256]  67 MB   (h2 dead by then)
    float* g2 = Rmid;   // [64,64,64,128] 134 MB   (h1 dead by then)
    float* g3 = Rbig;   // [64,128,128,64] 268 MB

    float* recon    = (float*)d_out;                 // [64,128,128,1]
    float* loss_out = (float*)d_out + 1048576;       // scalar
    int*   idx_out  = (int*)d_out + 1048577;         // [64,16,16]

    zero_accum_kernel<<<1, 1, 0, stream>>>(accum);

    // encoder stage 1 (CI=1: VALU)
    e1_kernel<<<32768, 256, 0, stream>>>(x, e1w, e1b, g1p, b1p, m1p, v1p, h1);

    // blocks = (CO/32) * ((64*OH*OW/16)/8);  LDS = KH*KW*(CI/32)*2048 bytes
    // e2: conv 4x4 s2 pad1, 32->32, 64 -> 32
    conv_wmma_kernel<64,64,32, 32,32,32, 4,4,2,1, 0,0,1>
        <<<512, 256, 32768, stream>>>(h1, e2w, e2b, g2p, b2p, m2p, v2p, h2);

    // e3: conv 3x3 s2 pad0, 32->32, 32 -> 16
    conv_wmma_kernel<32,32,32, 16,16,32, 3,3,2,0, 0,0,1>
        <<<128, 256, 18432, stream>>>(h2, e3w, e3b, g3p, b3p, m3p, v3p, h3);

    // e4: conv 1x1 s1, 32->64, relu only
    conv_wmma_kernel<16,16,32, 16,16,64, 1,1,1,0, 0,0,0>
        <<<256, 256, 2048, stream>>>(h3, e4w, e4b, nullptr, nullptr, nullptr, nullptr, z_e);

    // vector quantizer
    vq_kernel<<<64, 256, 0, stream>>>(z_e, emb, z_q, idx_out, accum);

    // decoder (conv-transpose SAME s2 k3 -> deconv pad 1), leaky-relu + BN
    conv_wmma_kernel<16,16,64, 32,32,256, 3,3,2,1, 1,1,1>
        <<<4096, 256, 36864, stream>>>(z_q, d1w, d1b, dg1, db1, dm1, dv1, g1);

    conv_wmma_kernel<32,32,256, 64,64,128, 3,3,2,1, 1,1,1>
        <<<8192, 256, 147456, stream>>>(g1, d2w, d2b, dg2, db2, dm2, dv2, g2);

    conv_wmma_kernel<64,64,128, 128,128,64, 3,3,2,1, 1,1,1>
        <<<16384, 256, 73728, stream>>>(g2, d3w, d3b, dg3, db3, dm3, dv3, g3);

    // d4: 1x1 64->1 + double sigmoid + SSE
    d4_kernel<<<4096, 256, 0, stream>>>(g3, d4w, d4b, x, recon, accum);

    finalize_kernel<<<1, 1, 0, stream>>>(accum, loss_out);
}
